// MoEMiniGPT_46789373723377
// MI455X (gfx1250) — matrix operand order repaired and checked by reading, not compiled, not run on hardware
//
#include <hip/hip_runtime.h>

// MoE top-2 routed FFN for MI455X (gfx1250, wave32, WMMA bf16).
// Needs ~82 MB of workspace (x in bf16 + pre-swizzled bf16 W1/W2 + routing lists).

#define N_TOK 4096   // B*S
#define D_DIM 768
#define F_DIM 3072
#define E_NUM 8
#define MT    32     // tokens per FFN tile (2 x 16-row WMMA M-halves)

typedef __attribute__((ext_vector_type(16))) __bf16 v16bf;
typedef __attribute__((ext_vector_type(8)))  __bf16 v8bf;
typedef __attribute__((ext_vector_type(4)))  __bf16 v4bf;
typedef __attribute__((ext_vector_type(8)))  float  v8f;

static __device__ __forceinline__ v16bf cat16(v8bf lo, v8bf hi) {
  return __builtin_shufflevector(lo, hi, 0,1,2,3,4,5,6,7,8,9,10,11,12,13,14,15);
}

static __device__ __forceinline__ v8f wmma_bf16(v16bf a, v16bf b, v8f c) {
  return __builtin_amdgcn_wmma_f32_16x16x32_bf16(false, a, false, b, (short)0, c,
                                                 false, false);
}

// ---------------------------------------------------------------------------
// 1) Router: logits -> softmax -> top-2 -> renormalized weights. 1 wave/token.
// ---------------------------------------------------------------------------
__global__ __launch_bounds__(256)
void router_kernel(const float* __restrict__ x, const float* __restrict__ wr,
                   int* __restrict__ tok_e, float* __restrict__ tok_w)
{
  __shared__ float swr[D_DIM * E_NUM];
  for (int i = threadIdx.x; i < D_DIM * E_NUM; i += 256) swr[i] = wr[i];
  __syncthreads();

  const int wave = threadIdx.x >> 5, lane = threadIdx.x & 31;
  const int t = blockIdx.x * 8 + wave;

  float acc[E_NUM];
  #pragma unroll
  for (int e = 0; e < E_NUM; ++e) acc[e] = 0.f;

  const float* xr = x + (size_t)t * D_DIM;
  for (int d = lane; d < D_DIM; d += 32) {
    const float xv = xr[d];
    #pragma unroll
    for (int e = 0; e < E_NUM; ++e) acc[e] = fmaf(xv, swr[d * E_NUM + e], acc[e]);
  }
  #pragma unroll
  for (int e = 0; e < E_NUM; ++e) {
    #pragma unroll
    for (int off = 16; off >= 1; off >>= 1) acc[e] += __shfl_xor(acc[e], off, 32);
  }

  if (lane == 0) {
    float m = acc[0];
    #pragma unroll
    for (int e = 1; e < E_NUM; ++e) m = fmaxf(m, acc[e]);
    float p[E_NUM], s = 0.f;
    #pragma unroll
    for (int e = 0; e < E_NUM; ++e) { p[e] = __expf(acc[e] - m); s += p[e]; }
    const float inv = 1.f / s;
    #pragma unroll
    for (int e = 0; e < E_NUM; ++e) p[e] *= inv;
    int i0 = 0;
    #pragma unroll
    for (int e = 1; e < E_NUM; ++e) if (p[e] > p[i0]) i0 = e;   // ties -> lowest idx
    int i1 = (i0 == 0) ? 1 : 0;
    #pragma unroll
    for (int e = 0; e < E_NUM; ++e) if (e != i0 && p[e] > p[i1]) i1 = e;
    const float den = 1.f / (p[i0] + p[i1] + 1e-8f);
    tok_e[t * 2]     = i0;  tok_w[t * 2]     = p[i0] * den;
    tok_e[t * 2 + 1] = i1;  tok_w[t * 2 + 1] = p[i1] * den;
  }
}

// ---------------------------------------------------------------------------
// 2) x fp32 -> bf16 (4-wide)
// ---------------------------------------------------------------------------
__global__ __launch_bounds__(256)
void convert_x_kernel(const float* __restrict__ x, __bf16* __restrict__ xbf)
{
  const int i = blockIdx.x * 256 + threadIdx.x;   // exact multiple, no bound check
  const float4 v = ((const float4*)x)[i];
  v4bf o;
  o[0] = (__bf16)v.x; o[1] = (__bf16)v.y; o[2] = (__bf16)v.z; o[3] = (__bf16)v.w;
  ((v4bf*)xbf)[i] = o;
}

// ---------------------------------------------------------------------------
// 3) Weight swizzle to the WMMA B-operand per-lane layout (K32 x N16 tiles):
//    lanes 0-15 : col n,    K = kbase + {0..7, 16..23}
//    lanes 16-31: col n-16, K = kbase + {8..15, 24..31}
//    Tile = 512 bf16 stored as [lane][16] contiguous -> two b128 loads/lane.
// ---------------------------------------------------------------------------
__global__ __launch_bounds__(512)
void swizzle_w1_kernel(const float* __restrict__ w1, __bf16* __restrict__ w1s)
{
  const int i = threadIdx.x & 15, lane = threadIdx.x >> 4;
  const int kT = blockIdx.x, fT = blockIdx.y, e = blockIdx.z;  // 24 x 192 x 8
  const int k = kT * 32 + (i & 7) + ((i >= 8) ? 16 : 0) + ((lane >= 16) ? 8 : 0);
  const int f = fT * 16 + (lane & 15);
  const size_t dst = ((((size_t)e * (F_DIM / 16) + fT) * (D_DIM / 32) + kT) * 512) + threadIdx.x;
  w1s[dst] = (__bf16)w1[((size_t)e * D_DIM + k) * F_DIM + f];
}

__global__ __launch_bounds__(512)
void swizzle_w2_kernel(const float* __restrict__ w2, __bf16* __restrict__ w2s)
{
  const int i = threadIdx.x & 15, lane = threadIdx.x >> 4;
  const int kT = blockIdx.x, nT = blockIdx.y, e = blockIdx.z;  // 96 x 48 x 8
  const int k = kT * 32 + (i & 7) + ((i >= 8) ? 16 : 0) + ((lane >= 16) ? 8 : 0);
  const int n = nT * 16 + (lane & 15);
  const size_t dst = ((((size_t)e * (D_DIM / 16) + nT) * (F_DIM / 32) + kT) * 512) + threadIdx.x;
  w2s[dst] = (__bf16)w2[((size_t)e * F_DIM + k) * D_DIM + n];
}

// ---------------------------------------------------------------------------
// 4) Stable per-expert token lists (deterministic: ballot prefix, no atomics)
// ---------------------------------------------------------------------------
__global__ __launch_bounds__(256)
void placement_kernel(const int* __restrict__ tok_e, const float* __restrict__ tok_w,
                      int* __restrict__ list_tok, float* __restrict__ list_w,
                      int* __restrict__ counts)
{
  const int e = blockIdx.x;
  __shared__ int woff[8];
  __shared__ int sbase;
  const int tid = threadIdx.x, wave = tid >> 5, lane = tid & 31;
  if (tid == 0) sbase = 0;
  __syncthreads();

  for (int t0 = 0; t0 < N_TOK; t0 += 256) {
    const int t = t0 + tid;
    const int e0 = tok_e[t * 2], e1 = tok_e[t * 2 + 1];
    const bool flag = (e0 == e) || (e1 == e);
    const float w = (e0 == e) ? tok_w[t * 2] : tok_w[t * 2 + 1];
    const unsigned mask = (unsigned)__ballot(flag);
    const int rank = __popc(mask & ((1u << lane) - 1u));
    if (lane == 0) woff[wave] = __popc(mask);
    __syncthreads();
    int wbase = sbase;
    for (int i = 0; i < wave; ++i) wbase += woff[i];
    if (flag) {
      const int pos = wbase + rank;
      list_tok[e * N_TOK + pos] = t;
      list_w[e * N_TOK + pos] = w;
    }
    __syncthreads();
    if (tid == 0) {
      int tot = 0;
      for (int i = 0; i < 8; ++i) tot += woff[i];
      sbase += tot;
    }
    __syncthreads();
  }
  if (tid == 0) counts[e] = sbase;
}

// ---------------------------------------------------------------------------
// 5) Fused routed FFN: per block = one expert x one 32-token tile.
//    Stage 1: H = gelu(X @ W1 + b1) per 128-wide F chunk
//             (8 waves x 16 cols, 2 M-halves each, B operand reused)
//    Stage 2: Y += H @ W2 (each wave owns 6 of 48 output N-tiles x 2 M-halves)
// ---------------------------------------------------------------------------
__global__ __launch_bounds__(256)
void moe_ffn_kernel(const __bf16* __restrict__ xbf, const __bf16* __restrict__ w1s,
                    const __bf16* __restrict__ w2s, const float* __restrict__ b1,
                    const float* __restrict__ b2, const int* __restrict__ list_tok,
                    const float* __restrict__ list_w, const int* __restrict__ counts,
                    float* __restrict__ out)
{
  const int e = blockIdx.y;
  const int cnt = counts[e];
  const int m0 = blockIdx.x * MT;
  if (m0 >= cnt) return;     // uniform per block

  __shared__ __align__(16) __bf16 Xt[MT * D_DIM];   // 48 KB
  __shared__ __align__(16) __bf16 Ht[MT * 128];     // 8 KB
  __shared__ int   s_tok[MT];
  __shared__ float s_wt[MT];

  const int tid  = threadIdx.x;
  const int wave = tid >> 5;
  const int lane = tid & 31;
  const int lrow = lane & 15;
  const int lhi  = lane >> 4;

  if (tid < MT) {
    const int g = m0 + tid;
    s_tok[tid] = (g < cnt) ? list_tok[e * N_TOK + g] : 0;
    s_wt[tid]  = (g < cnt) ? list_w[e * N_TOK + g]   : 0.f;  // padded rows contribute 0
  }
  __syncthreads();

  // Gather MT token rows (bf16) into LDS via CDNA5 async LDS-DMA loads
  // (ASYNCcnt-tracked; VDST = LDS byte offset = low 32b of flat LDS address).
  for (int c = tid; c < MT * (D_DIM / 8); c += 256) {
    const int r = c / (D_DIM / 8), o = (c % (D_DIM / 8)) * 8;
    const unsigned long long gaddr =
        (unsigned long long)(size_t)(xbf + (size_t)s_tok[r] * D_DIM + o);
    const unsigned laddr = (unsigned)(size_t)(&Xt[r * D_DIM + o]);
    asm volatile("global_load_async_to_lds_b128 %0, %1, off"
                 :: "v"(laddr), "v"(gaddr) : "memory");
  }
  asm volatile("s_wait_asynccnt 0" ::: "memory");

  v8f yacc[12];                        // 6 N-tiles x 2 M-halves
  #pragma unroll
  for (int j = 0; j < 12; ++j)
    #pragma unroll
    for (int r = 0; r < 8; ++r) yacc[j][r] = 0.f;

  __syncthreads();

  for (int fc = 0; fc < F_DIM / 128; ++fc) {
    // ---- stage 1: this wave's 16 H columns x 2 M-halves, reduce over D ----
    const float bias = b1[e * F_DIM + fc * 128 + wave * 16 + lrow];
    v8f hacc0, hacc1;
    #pragma unroll
    for (int r = 0; r < 8; ++r) { hacc0[r] = bias; hacc1[r] = bias; }

    const __bf16* w1tile =
        w1s + ((size_t)(e * (F_DIM / 16) + (fc * 8 + wave))) * (D_DIM / 32) * 512;
    #pragma unroll 2
    for (int kT = 0; kT < D_DIM / 32; ++kT) {
      // A: 16x32 bf16 (lane<16: row=lane, K {0..7,16..23}; lane>=16: K {8..15,24..31})
      const __bf16* ap0 = &Xt[lrow * D_DIM + kT * 32 + lhi * 8];
      const __bf16* ap1 = ap0 + 16 * D_DIM;
      const __bf16* bp  = w1tile + kT * 512 + lane * 16;
      const v16bf Bm = cat16(*(const v8bf*)bp, *(const v8bf*)(bp + 8));
      const v16bf A0 = cat16(*(const v8bf*)ap0, *(const v8bf*)(ap0 + 16));
      const v16bf A1 = cat16(*(const v8bf*)ap1, *(const v8bf*)(ap1 + 16));
      hacc0 = wmma_bf16(A0, Bm, hacc0);
      hacc1 = wmma_bf16(A1, Bm, hacc1);
    }

    __syncthreads();  // previous iteration's stage-2 reads of Ht complete
    {
      const int col = wave * 16 + lrow;
      const int rbase = lhi * 8;
      #pragma unroll
      for (int r = 0; r < 8; ++r) {
        float h0 = hacc0[r];
        h0 = 0.5f * h0 * (1.0f + erff(h0 * 0.70710678118654752f));  // exact gelu
        Ht[(rbase + r) * 128 + col] = (__bf16)h0;
        float h1 = hacc1[r];
        h1 = 0.5f * h1 * (1.0f + erff(h1 * 0.70710678118654752f));
        Ht[(16 + rbase + r) * 128 + col] = (__bf16)h1;
      }
    }
    __syncthreads();

    // ---- stage 2: Y += Ht(32x128) @ W2(128x768), 6 N-tiles x 2 halves / wave ----
    #pragma unroll
    for (int j = 0; j < 6; ++j) {
      const int nt = wave * 6 + j;
      const __bf16* w2tile =
          w2s + (((size_t)(e * (D_DIM / 16) + nt)) * (F_DIM / 32) + fc * 4) * 512;
      #pragma unroll
      for (int kk = 0; kk < 4; ++kk) {
        const __bf16* bp = w2tile + kk * 512 + lane * 16;
        const v16bf Bm = cat16(*(const v8bf*)bp, *(const v8bf*)(bp + 8));
        const __bf16* ap0 = &Ht[lrow * 128 + kk * 32 + lhi * 8];
        const __bf16* ap1 = ap0 + 16 * 128;
        const v16bf A0 = cat16(*(const v8bf*)ap0, *(const v8bf*)(ap0 + 16));
        const v16bf A1 = cat16(*(const v8bf*)ap1, *(const v8bf*)(ap1 + 16));
        yacc[j * 2]     = wmma_bf16(A0, Bm, yacc[j * 2]);
        yacc[j * 2 + 1] = wmma_bf16(A1, Bm, yacc[j * 2 + 1]);
      }
    }
  }

  // Epilogue: out[token] += weight * (Y + b2). Exactly 2 fp32 contributions per
  // element; 2-term fp32 add is commutative => bitwise deterministic.
  #pragma unroll
  for (int j = 0; j < 6; ++j) {
    const int nt = wave * 6 + j;
    const int d = nt * 16 + lrow;
    const float bb = b2[e * D_DIM + d];
    const int rbase = lhi * 8;
    #pragma unroll
    for (int h = 0; h < 2; ++h) {
      #pragma unroll
      for (int r = 0; r < 8; ++r) {
        const int m = h * 16 + rbase + r;
        atomicAdd(&out[(size_t)s_tok[m] * D_DIM + d], s_wt[m] * (yacc[j * 2 + h][r] + bb));
      }
    }
  }
}

// ---------------------------------------------------------------------------
extern "C" void kernel_launch(void* const* d_in, const int* in_sizes, int n_in,
                              void* d_out, int out_size, void* d_ws, size_t ws_size,
                              hipStream_t stream)
{
  const float* x  = (const float*)d_in[0];
  const float* wr = (const float*)d_in[1];
  const float* w1 = (const float*)d_in[2];
  const float* b1 = (const float*)d_in[3];
  const float* w2 = (const float*)d_in[4];
  const float* b2 = (const float*)d_in[5];
  float* out = (float*)d_out;

  char* ws = (char*)d_ws;
  size_t off = 0;
  auto take = [&](size_t bytes) {
    char* p = ws + off;
    off = (off + bytes + 255) & ~(size_t)255;
    return p;
  };
  int*    tok_e    = (int*)   take((size_t)N_TOK * 2 * sizeof(int));
  float*  tok_w    = (float*) take((size_t)N_TOK * 2 * sizeof(float));
  int*    list_tok = (int*)   take((size_t)E_NUM * N_TOK * sizeof(int));
  float*  list_w   = (float*) take((size_t)E_NUM * N_TOK * sizeof(float));
  int*    counts   = (int*)   take((size_t)E_NUM * sizeof(int));
  __bf16* xbf      = (__bf16*)take((size_t)N_TOK * D_DIM * 2);
  __bf16* w1s      = (__bf16*)take((size_t)E_NUM * D_DIM * F_DIM * 2);
  __bf16* w2s      = (__bf16*)take((size_t)E_NUM * D_DIM * F_DIM * 2);
  (void)ws_size; (void)in_sizes; (void)n_in;

  (void)hipMemsetAsync(d_out, 0, (size_t)out_size * sizeof(float), stream);

  router_kernel<<<N_TOK / 8, 256, 0, stream>>>(x, wr, tok_e, tok_w);
  convert_x_kernel<<<(N_TOK * D_DIM / 4) / 256, 256, 0, stream>>>(x, xbf);
  swizzle_w1_kernel<<<dim3(D_DIM / 32, F_DIM / 16, E_NUM), 512, 0, stream>>>(w1, w1s);
  swizzle_w2_kernel<<<dim3(F_DIM / 32, D_DIM / 16, E_NUM), 512, 0, stream>>>(w2, w2s);
  placement_kernel<<<E_NUM, 256, 0, stream>>>(tok_e, tok_w, list_tok, list_w, counts);
  moe_ffn_kernel<<<dim3(N_TOK / MT, E_NUM), 256, 0, stream>>>(
      xbf, w1s, w2s, b1, b2, list_tok, list_w, counts, out);
}
